// MLPConceptEmbedder_53154515255623
// MI455X (gfx1250) — compile-verified, matrix operand order checked by reading, not run on hardware
//
#include <hip/hip_runtime.h>
#include <stdint.h>

#define B_    8
#define T_    2048
#define N_    16
#define DIN   1024
#define HH    512
#define DOUT  128
#define M_TOT (B_ * T_)        // 16384 tokens

#define M_TILE 64              // tokens per workgroup
#define HC     64              // H columns per fused chunk
#define KSLAB  64              // K staged per LDS slab (2 WMMA k-steps)
#define NSLAB  (DIN / KSLAB)   // 16 slabs per chunk

#define XS_STRIDE 1032         // bf16 elems per x row in LDS (+8 pad)
#define S_STRIDE  72           // bf16 elems per slab row (+8 pad)

// LDS byte offsets
#define XS_OFF   0
#define XS_BYTES (M_TILE * XS_STRIDE * 2)          // 132096
#define WS1_OFF  (XS_OFF + XS_BYTES)
#define WS1_BUF  (HC * S_STRIDE * 2)               // 9216 per buffer
#define HS_OFF   (WS1_OFF + 2 * WS1_BUF)
#define HS_BYTES (M_TILE * S_STRIDE * 2)           // 9216
#define W2_OFF   (HS_OFF + HS_BYTES)
#define W2_BYTES (DOUT * S_STRIDE * 2)             // 18432
#define SMEM_BYTES (W2_OFF + W2_BYTES)             // 178176

typedef __bf16 v16bf __attribute__((ext_vector_type(16)));
typedef float  v8f   __attribute__((ext_vector_type(8)));
typedef int    v4i   __attribute__((ext_vector_type(4)));

#define AS1 __attribute__((address_space(1)))
#define AS3 __attribute__((address_space(3)))

__device__ __forceinline__ unsigned short f2bf(float f) {
  union { float f; uint32_t u; } c; c.f = f;
  uint32_t r = c.u + 0x7FFFu + ((c.u >> 16) & 1u);   // round-to-nearest-even
  return (unsigned short)(r >> 16);
}

// 16-byte global -> LDS copy on the CDNA5 async path (ASYNCcnt),
// with a synchronous VGPR fallback if the builtin is unavailable.
__device__ __forceinline__ void async_cp16(void* lds_dst, const void* gsrc) {
#if __has_builtin(__builtin_amdgcn_global_load_async_to_lds_b128)
  __builtin_amdgcn_global_load_async_to_lds_b128(
      (AS1 v4i*)(uintptr_t)gsrc,
      (AS3 v4i*)(uint32_t)(uintptr_t)lds_dst, 0, 0);
#else
  *(uint4*)lds_dst = *(const uint4*)gsrc;
#endif
}

__device__ __forceinline__ void async_wait_all() {
#if __has_builtin(__builtin_amdgcn_s_wait_asynccnt)
  __builtin_amdgcn_s_wait_asynccnt(0);
#elif __has_builtin(__builtin_amdgcn_global_load_async_to_lds_b128)
  asm volatile("s_wait_asynccnt 0" ::: "memory");
#endif
}

// 16 bf16 fragment = two ds_load_b128
__device__ __forceinline__ v16bf load_frag16(const unsigned short* p) {
  union { uint4 q[2]; v16bf v; } f;
  f.q[0] = *(const uint4*)(p);
  f.q[1] = *(const uint4*)(p + 16);
  return f.v;
}

// ---------------- prep pass: f32 -> bf16, weights pre-transposed ----------------

__global__ __launch_bounds__(256) void cvt_x_kernel(const float* __restrict__ x,
                                                    unsigned short* __restrict__ xb) {
  size_t i = (size_t)blockIdx.x * 256 + threadIdx.x;       // float4 index
  float4 v = *(const float4*)(x + i * 4);
  union { unsigned short s[4]; uint2 u; } pk;
  pk.s[0] = f2bf(v.x); pk.s[1] = f2bf(v.y); pk.s[2] = f2bf(v.z); pk.s[3] = f2bf(v.w);
  *(uint2*)(xb + i * 4) = pk.u;
}

// W1 [n][k][h] (f32) -> W1t [n][h][k] (bf16)
__global__ __launch_bounds__(256) void cvt_w1_kernel(const float* __restrict__ W1,
                                                     unsigned short* __restrict__ W1t) {
  size_t i = (size_t)blockIdx.x * 256 + threadIdx.x;       // float4 index
  int n   = (int)(i >> 17);                                 // 131072 float4 per concept
  int rem = (int)(i & 131071);
  int k   = rem >> 7;                                       // 128 float4 per k row
  int h4  = rem & 127;
  float4 v = *(const float4*)(W1 + ((size_t)n << 19) + (size_t)k * HH + h4 * 4);
  size_t base = ((size_t)n * HH + h4 * 4) * DIN + k;
  W1t[base            ] = f2bf(v.x);
  W1t[base + 1 * DIN  ] = f2bf(v.y);
  W1t[base + 2 * DIN  ] = f2bf(v.z);
  W1t[base + 3 * DIN  ] = f2bf(v.w);
}

// W2 [n][h][o] (f32) -> W2t [n][o][h] (bf16)
__global__ __launch_bounds__(256) void cvt_w2_kernel(const float* __restrict__ W2,
                                                     unsigned short* __restrict__ W2t) {
  size_t i = (size_t)blockIdx.x * 256 + threadIdx.x;       // float4 index
  int n   = (int)(i >> 14);                                 // 16384 float4 per concept
  int rem = (int)(i & 16383);
  int h   = rem >> 5;                                       // 32 float4 per h row
  int o4  = rem & 31;
  float4 v = *(const float4*)(W2 + ((size_t)n << 16) + (size_t)h * DOUT + o4 * 4);
  size_t base = ((size_t)n * DOUT + o4 * 4) * HH + h;
  W2t[base           ] = f2bf(v.x);
  W2t[base + 1 * HH  ] = f2bf(v.y);
  W2t[base + 2 * HH  ] = f2bf(v.z);
  W2t[base + 3 * HH  ] = f2bf(v.w);
}

// ---------------- fused grouped-MLP WMMA kernel ----------------

__global__ __launch_bounds__(256, 1)
void mlp_concept_kernel(const unsigned short* __restrict__ xb,    // [M][1024] bf16
                        const unsigned short* __restrict__ W1t,   // [n][h][k] bf16
                        const unsigned short* __restrict__ W2t,   // [n][o][h] bf16
                        const float* __restrict__ b1,
                        const float* __restrict__ b2p,
                        float* __restrict__ out) {
  extern __shared__ unsigned char smem[];
  unsigned char* xs_b  = smem + XS_OFF;
  unsigned char* ws1_b = smem + WS1_OFF;
  unsigned char* hs_b  = smem + HS_OFF;
  unsigned char* w2_b  = smem + W2_OFF;
  const unsigned short* xs_u  = (const unsigned short*)xs_b;
  const unsigned short* hs_cu = (const unsigned short*)hs_b;
  unsigned short* hs_u = (unsigned short*)hs_b;
  const unsigned short* w2_u  = (const unsigned short*)w2_b;

  const int tid  = threadIdx.x;
  const int wave = tid >> 5;
  const int lane = tid & 31;
  const int l16  = lane & 15;
  const int g    = lane >> 4;            // K-half group per WMMA layout

  const int m0 = blockIdx.x * M_TILE;
  const int n  = blockIdx.y;

  const unsigned char* xb8  = (const unsigned char*)xb + (size_t)m0 * DIN * 2;
  const unsigned char* w1n8 = (const unsigned char*)W1t + (size_t)n * HH * DIN * 2;
  const unsigned char* w2n8 = (const unsigned char*)W2t + (size_t)n * DOUT * HH * 2;

  // ---- async-stage the whole bf16 x tile (resident across all 8 H-chunks)
  #pragma unroll
  for (int i = 0; i < 32; ++i) {                 // 8192 x 16B chunks / 256 threads
    int idx = tid + i * 256;
    int r   = idx >> 7;                          // 128 chunks per token row
    int c   = idx & 127;
    async_cp16(xs_b + r * (XS_STRIDE * 2) + c * 16,
               xb8 + (size_t)r * (DIN * 2) + c * 16);
  }

  const int wrow = (wave >> 1) * 16;
  const int wn1  = (wave & 1) * 32;              // layer1: 2 tiles of 16 cols
  const int wn2  = (wave & 1) * 64;              // layer2: 4 tiles of 16 cols

  v8f oacc[4] = {};

  for (int hc = 0; hc < HH; hc += HC) {
    __syncthreads();                             // prev chunk done with hs / w2t / slabs

    // issue W2 chunk copy: [128 o][64 h] (contiguous along h)
    #pragma unroll
    for (int i = 0; i < 4; ++i) {                // 1024 chunks
      int idx = tid + i * 256;
      int o   = idx >> 3;                        // 8 chunks per o row
      int c   = idx & 7;
      async_cp16(w2_b + o * (S_STRIDE * 2) + c * 16,
                 w2n8 + (size_t)o * (HH * 2) + hc * 2 + c * 16);
    }
    // issue first W1 slab: [64 h][64 k] (contiguous along k)
    #pragma unroll
    for (int i = 0; i < 2; ++i) {                // 512 chunks
      int idx = tid + i * 256;
      int h   = idx >> 3;
      int c   = idx & 7;
      async_cp16(ws1_b + h * (S_STRIDE * 2) + c * 16,
                 w1n8 + (size_t)(hc + h) * (DIN * 2) + c * 16);
    }

    v8f hacc[2] = {};

    // ---- layer 1: K=1024 reduction, double-buffered async slab pipeline
    for (int s = 0; s < NSLAB; ++s) {
      async_wait_all();                          // slab s (+ w2t/x on s==0) landed
      __syncthreads();                           // all waves landed; other buf free
      if (s + 1 < NSLAB) {                       // overlap DMA of slab s+1 with WMMA
        unsigned char* dst = ws1_b + ((s + 1) & 1) * WS1_BUF;
        int k0n = (s + 1) * KSLAB;
        #pragma unroll
        for (int i = 0; i < 2; ++i) {
          int idx = tid + i * 256;
          int h   = idx >> 3;
          int c   = idx & 7;
          async_cp16(dst + h * (S_STRIDE * 2) + c * 16,
                     w1n8 + (size_t)(hc + h) * (DIN * 2) + k0n * 2 + c * 16);
        }
      }
      const unsigned short* wsu = (const unsigned short*)(ws1_b + (s & 1) * WS1_BUF);
      const int k0 = s * KSLAB;
      #pragma unroll
      for (int ks = 0; ks < KSLAB; ks += 32) {
        v16bf a = load_frag16(&xs_u[(wrow + l16) * XS_STRIDE + k0 + ks + g * 8]);
        #pragma unroll
        for (int t = 0; t < 2; ++t) {
          v16bf b = load_frag16(&wsu[(wn1 + t * 16 + l16) * S_STRIDE + ks + g * 8]);
          hacc[t] = __builtin_amdgcn_wmma_f32_16x16x32_bf16(
              false, a, false, b, (short)0, hacc[t], false, false);
        }
      }
    }

    // ---- bias + ReLU -> bf16 h chunk in LDS (never touches global)
    #pragma unroll
    for (int t = 0; t < 2; ++t) {
      int col = wn1 + t * 16 + l16;
      float bias = b1[n * HH + hc + col];
      #pragma unroll
      for (int r = 0; r < 8; ++r) {
        float v = hacc[t][r] + bias;
        v = v > 0.0f ? v : 0.0f;
        hs_u[(wrow + g * 8 + r) * S_STRIDE + col] = f2bf(v);
      }
    }
    __syncthreads();                             // hs + w2t visible

    // ---- layer 2: oacc += relu(h_chunk) @ W2_chunk (K = 64)
    #pragma unroll
    for (int ks = 0; ks < HC; ks += 32) {
      v16bf a2 = load_frag16(&hs_cu[(wrow + l16) * S_STRIDE + ks + g * 8]);
      #pragma unroll
      for (int t = 0; t < 4; ++t) {
        v16bf bw = load_frag16(&w2_u[(wn2 + t * 16 + l16) * S_STRIDE + ks + g * 8]);
        oacc[t] = __builtin_amdgcn_wmma_f32_16x16x32_bf16(
            false, a2, false, bw, (short)0, oacc[t], false, false);
      }
    }
  }

  // ---- epilogue: + b2, f32 store out[m, n, o]
  #pragma unroll
  for (int t = 0; t < 4; ++t) {
    int col = wn2 + t * 16 + l16;
    float bias = b2p[n * DOUT + col];
    #pragma unroll
    for (int r = 0; r < 8; ++r) {
      int m = m0 + wrow + g * 8 + r;
      out[(size_t)m * (N_ * DOUT) + n * DOUT + col] = oacc[t][r] + bias;
    }
  }
}

extern "C" void kernel_launch(void* const* d_in, const int* in_sizes, int n_in,
                              void* d_out, int out_size, void* d_ws, size_t ws_size,
                              hipStream_t stream) {
  const float* x  = (const float*)d_in[0];
  const float* W1 = (const float*)d_in[1];
  const float* b1 = (const float*)d_in[2];
  const float* W2 = (const float*)d_in[3];
  const float* b2 = (const float*)d_in[4];
  float* out = (float*)d_out;

  // workspace layout (bf16): x copy, W1 transposed, W2 transposed  (~50 MB)
  unsigned short* xb  = (unsigned short*)d_ws;                          // 16384*1024
  unsigned short* W1t = xb  + (size_t)M_TOT * DIN;                      // 16*512*1024
  unsigned short* W2t = W1t + (size_t)N_ * HH * DIN;                    // 16*128*512

  cvt_x_kernel <<<(M_TOT * DIN / 4) / 256, 256, 0, stream>>>(x, xb);
  cvt_w1_kernel<<<(N_ * DIN * HH / 4) / 256, 256, 0, stream>>>(W1, W1t);
  cvt_w2_kernel<<<(N_ * HH * DOUT / 4) / 256, 256, 0, stream>>>(W2, W2t);

  hipError_t attr_rc = hipFuncSetAttribute(
      reinterpret_cast<const void*>(mlp_concept_kernel),
      hipFuncAttributeMaxDynamicSharedMemorySize, (int)SMEM_BYTES);
  (void)attr_rc;

  dim3 grid(M_TOT / M_TILE, N_);                 // 256 token tiles x 16 concepts
  mlp_concept_kernel<<<grid, 256, SMEM_BYTES, stream>>>(xb, W1t, W2t, b1, b2, out);
}
